// MoET5Block_86698209837683
// MI455X (gfx1250) — matrix operand order repaired and checked by reading, not compiled, not run on hardware
//
#include <hip/hip_runtime.h>

typedef __bf16 bf16;
typedef __attribute__((ext_vector_type(2)))  __bf16 bf16x2;
typedef __attribute__((ext_vector_type(4)))  __bf16 bf16x4;
typedef __attribute__((ext_vector_type(8)))  __bf16 bf16x8;
typedef __attribute__((ext_vector_type(16))) __bf16 bf16x16;
typedef __attribute__((ext_vector_type(8)))  float  f32x8;

#define DEV static __device__ __forceinline__

#define BSZ    8
#define SEQ    1024
#define DMODEL 1024
#define NHEADS 16
#define HDIM   64
#define NEXP   16
#define CAP    128
#define DHID   2730
#define NTOK   (BSZ*SEQ)

#define BTS 40   // padded LDS stride (elems) for 32-wide K tiles
#define KTS 72   // padded LDS stride (elems) for 64-wide tiles

DEV f32x8 zero8() {
  f32x8 z;
#pragma unroll
  for (int i = 0; i < 8; i++) z[i] = 0.f;
  return z;
}

DEV f32x8 wmma_bf16(bf16x16 a, bf16x16 b, f32x8 c) {
  return __builtin_amdgcn_wmma_f32_16x16x32_bf16(false, a, false, b, (short)0, c,
                                                 false, false);
}

// A-fragment (16x32 bf16): lane L: row = L&15, elems K = h*16 + {0..15}
DEV bf16x16 load_a_frag(const bf16* p) {
  bf16x8 lo = *(const bf16x8*)(p);
  bf16x8 hi = *(const bf16x8*)(p + 8);
  bf16x16 a;
#pragma unroll
  for (int i = 0; i < 8; i++) { a[i] = lo[i]; a[i + 8] = hi[i]; }
  return a;
}

// B-fragment from Bt (NxK) layout: lane L: col = L&15, elems K = h*16 + {0..15}
DEV bf16x16 load_b_frag(const bf16* p) {
  bf16x8 lo = *(const bf16x8*)(p);
  bf16x8 hi = *(const bf16x8*)(p + 8);
  bf16x16 b;
#pragma unroll
  for (int i = 0; i < 8; i++) { b[i] = lo[i]; b[i + 8] = hi[i]; }
  return b;
}

DEV float gelu_tanh(float x) {
  float x3 = x * x * x;
  return 0.5f * x * (1.f + tanhf(0.7978845608028654f * (x + 0.044715f * x3)));
}

// ---------------------------------------------------------------- converter
__global__ void k_f32_to_bf16(const float* __restrict__ s, bf16* __restrict__ d,
                              long long n) {
  long long i = (long long)blockIdx.x * blockDim.x + threadIdx.x;
  long long st = (long long)gridDim.x * blockDim.x;
  long long n4 = n >> 2;
  for (long long j = i; j < n4; j += st) {
    float4 v = ((const float4*)s)[j];
    bf16x4 h;
    h[0] = (bf16)v.x; h[1] = (bf16)v.y; h[2] = (bf16)v.z; h[3] = (bf16)v.w;
    ((bf16x4*)d)[j] = h;
  }
  if (i == 0)
    for (long long j = n4 << 2; j < n; j++) d[j] = (bf16)s[j];
}

// ------------------------------------------------------------------- norms
// mode 0: T5 layernorm, mode 1: st-moe RMSNorm. One float4 per thread.
__global__ __launch_bounds__(256)
void k_norm(const float* __restrict__ x, const float* __restrict__ w,
            float* __restrict__ outF, bf16* __restrict__ outH, int mode) {
  __shared__ float red[256];
  const int row = blockIdx.x, tid = threadIdx.x;
  float4 xv = ((const float4*)(x + (size_t)row * DMODEL))[tid];
  float s = xv.x * xv.x + xv.y * xv.y + xv.z * xv.z + xv.w * xv.w;
  red[tid] = s;
  __syncthreads();
  for (int st = 128; st > 0; st >>= 1) {
    if (tid < st) red[tid] += red[tid + st];
    __syncthreads();
  }
  float tot = red[0];
  float scale = (mode == 0) ? rsqrtf(tot * (1.f / DMODEL) + 1e-6f)
                            : (32.f / fmaxf(sqrtf(tot), 1e-12f));
  float4 wv = ((const float4*)w)[tid];
  float4 o;
  o.x = xv.x * scale * wv.x; o.y = xv.y * scale * wv.y;
  o.z = xv.z * scale * wv.z; o.w = xv.w * scale * wv.w;
  if (outF) ((float4*)(outF + (size_t)row * DMODEL))[tid] = o;
  if (outH) {
    bf16x4 h;
    h[0] = (bf16)o.x; h[1] = (bf16)o.y; h[2] = (bf16)o.z; h[3] = (bf16)o.w;
    ((bf16x4*)(outH + (size_t)row * DMODEL))[tid] = h;
  }
}

// ------------------------------------------------------------------- GEMM
__global__ __launch_bounds__(256)
void k_gemm(const bf16* __restrict__ A, long long sAe, int lda,
            const bf16* __restrict__ Bm, long long sBe, int ldb,
            const float* __restrict__ bias, int sBiasE,
            const float* __restrict__ res0, const float* __restrict__ res1,
            float* __restrict__ outF, bf16* __restrict__ outH, long long sCe,
            int M, int N, int K, int ldc) {
  __shared__ __align__(16) bf16 lA[128 * BTS];
  __shared__ __align__(16) bf16 lB[128 * BTS];
  const int tid = threadIdx.x;
  const int z = blockIdx.z;
  const int m0 = blockIdx.y * 128, n0 = blockIdx.x * 128;
  const bf16* Ae = A + (size_t)z * sAe;
  const bf16* Be = Bm + (size_t)z * sBe;
  const int wv = tid >> 5, ln = tid & 31, rl = ln & 15, hf = ln >> 4;
  const bool ldaAligned = (lda & 7) == 0;
  const bool fullN = (n0 + 128 <= N);
  f32x8 acc[8];
#pragma unroll
  for (int i = 0; i < 8; i++) acc[i] = zero8();

  for (int k0 = 0; k0 < K; k0 += 32) {
    __syncthreads();
    if ((k0 + 32 <= K) && fullN) {               // fast path: whole tile in range
      if (ldaAligned) {
#pragma unroll
        for (int i = 0; i < 2; i++) {            // A tile: 16B chunks
          int cid = tid + i * 256;               // 0..511
          int r = cid >> 2, c = (cid & 3) * 8;
          *(bf16x8*)&lA[r * BTS + c] =
              *(const bf16x8*)(Ae + (size_t)(m0 + r) * lda + (k0 + c));
        }
      } else {
#pragma unroll
        for (int i = 0; i < 8; i++) {            // A tile: 4B chunks
          int cid = tid + i * 256;               // 0..2047
          int r = cid >> 4, c = (cid & 15) * 2;
          *(bf16x2*)&lA[r * BTS + c] =
              *(const bf16x2*)(Ae + (size_t)(m0 + r) * lda + (k0 + c));
        }
      }
#pragma unroll
      for (int i = 0; i < 4; i++) {              // B tile: 8B reads, transposed
        int cid = tid + i * 256;                 // 0..1023
        int r = cid >> 5, c = (cid & 31) * 4;
        bf16x4 bv = *(const bf16x4*)(Be + (size_t)(k0 + r) * ldb + (n0 + c));
#pragma unroll
        for (int j = 0; j < 4; j++) lB[(c + j) * BTS + r] = bv[j];
      }
      if (k0 + 32 < K) {
        __builtin_prefetch(Ae + (size_t)(m0 + (tid >> 1)) * lda + k0 + 32, 0, 1);
        __builtin_prefetch(Be + (size_t)(k0 + 32 + (tid & 31)) * ldb + n0, 0, 1);
      }
    } else {                                     // edge path (guarded, scalar)
#pragma unroll
      for (int i = 0; i < 16; i++) {
        int idx = tid + i * 256;
        int r = idx >> 5, c = idx & 31;
        int gk = k0 + c;
        lA[r * BTS + c] = (gk < K) ? Ae[(size_t)(m0 + r) * lda + gk] : (bf16)0.f;
      }
#pragma unroll
      for (int i = 0; i < 16; i++) {
        int idx = tid + i * 256;
        int r = idx >> 7, c = idx & 127;
        int gk = k0 + r, gn = n0 + c;
        lB[c * BTS + r] =
            (gk < K && gn < N) ? Be[(size_t)gk * ldb + gn] : (bf16)0.f;
      }
    }
    __syncthreads();
    bf16x16 af = load_a_frag(&lA[(wv * 16 + rl) * BTS + hf * 16]);
#pragma unroll
    for (int nt = 0; nt < 8; nt++) {
      bf16x16 bf_ = load_b_frag(&lB[(nt * 16 + rl) * BTS + hf * 16]);
      acc[nt] = wmma_bf16(af, bf_, acc[nt]);
    }
  }
#pragma unroll
  for (int nt = 0; nt < 8; nt++) {
    int n = n0 + nt * 16 + rl;
    if (n >= N) continue;
    float bv = bias ? bias[(size_t)z * sBiasE + n] : 0.f;
#pragma unroll
    for (int vr = 0; vr < 8; vr++) {
      int m = m0 + wv * 16 + hf * 8 + vr;
      float v = acc[nt][vr] + bv;
      size_t ri = (size_t)m * ldc + n;
      if (res0) v += res0[ri];
      if (res1) v += res1[ri];
      size_t oi = (size_t)z * sCe + ri;
      if (outF) outF[oi] = v;
      if (outH) outH[oi] = (bf16)v;
    }
  }
}

// ------------------------------------------------------------- GLU GEMM
__global__ __launch_bounds__(256)
void k_gemm_glu(const bf16* __restrict__ A, long long sAe, int lda,
                const bf16* __restrict__ Bm, long long sBe, int ldb,
                const float* __restrict__ bias, int sBiasE,
                const float* __restrict__ mult, int sMultE,
                bf16* __restrict__ outH, long long sHe,
                int M, int Hn, int K) {
  __shared__ __align__(16) bf16 lA[128 * BTS];
  __shared__ __align__(16) bf16 lBa[128 * BTS];
  __shared__ __align__(16) bf16 lBg[128 * BTS];
  const int tid = threadIdx.x;
  const int z = blockIdx.z;
  const int m0 = blockIdx.y * 128, n0 = blockIdx.x * 128;
  const bf16* Ae = A + (size_t)z * sAe;
  const bf16* Be = Bm + (size_t)z * sBe;
  const int wv = tid >> 5, ln = tid & 31, rl = ln & 15, hf = ln >> 4;
  const bool fullN = (n0 + 128 <= Hn);
  f32x8 accA[8], accG[8];
#pragma unroll
  for (int i = 0; i < 8; i++) { accA[i] = zero8(); accG[i] = zero8(); }

  for (int k0 = 0; k0 < K; k0 += 32) {
    __syncthreads();
    if ((k0 + 32 <= K) && fullN) {
#pragma unroll
      for (int i = 0; i < 2; i++) {              // A tile: 16B chunks (lda%8==0 here)
        int cid = tid + i * 256;
        int r = cid >> 2, c = (cid & 3) * 8;
        *(bf16x8*)&lA[r * BTS + c] =
            *(const bf16x8*)(Ae + (size_t)(m0 + r) * lda + (k0 + c));
      }
#pragma unroll
      for (int i = 0; i < 4; i++) {              // two B tiles, transposed
        int cid = tid + i * 256;
        int r = cid >> 5, c = (cid & 31) * 4;
        const bf16* bp = Be + (size_t)(k0 + r) * ldb + (n0 + c);
        bf16x4 ba = *(const bf16x4*)bp;          // 8B-aligned
        bf16x2 bg0 = *(const bf16x2*)(bp + Hn);  // +Hn breaks 8B align -> 4B
        bf16x2 bg1 = *(const bf16x2*)(bp + Hn + 2);
#pragma unroll
        for (int j = 0; j < 4; j++) lBa[(c + j) * BTS + r] = ba[j];
        lBg[(c + 0) * BTS + r] = bg0[0];
        lBg[(c + 1) * BTS + r] = bg0[1];
        lBg[(c + 2) * BTS + r] = bg1[0];
        lBg[(c + 3) * BTS + r] = bg1[1];
      }
      if (k0 + 32 < K) {
        __builtin_prefetch(Ae + (size_t)(m0 + (tid >> 1)) * lda + k0 + 32, 0, 1);
        __builtin_prefetch(Be + (size_t)(k0 + 32 + (tid & 31)) * ldb + n0, 0, 1);
      }
    } else {
#pragma unroll
      for (int i = 0; i < 16; i++) {
        int idx = tid + i * 256;
        int r = idx >> 5, c = idx & 31;
        int gk = k0 + c;
        lA[r * BTS + c] = (gk < K) ? Ae[(size_t)(m0 + r) * lda + gk] : (bf16)0.f;
      }
#pragma unroll
      for (int i = 0; i < 16; i++) {
        int idx = tid + i * 256;
        int r = idx >> 7, c = idx & 127;
        int gk = k0 + r, gn = n0 + c;
        bool ok = (gk < K) && (gn < Hn);
        lBa[c * BTS + r] = ok ? Be[(size_t)gk * ldb + gn] : (bf16)0.f;
        lBg[c * BTS + r] = ok ? Be[(size_t)gk * ldb + gn + Hn] : (bf16)0.f;
      }
    }
    __syncthreads();
    bf16x16 af = load_a_frag(&lA[(wv * 16 + rl) * BTS + hf * 16]);
#pragma unroll
    for (int nt = 0; nt < 8; nt++) {
      bf16x16 ba = load_b_frag(&lBa[(nt * 16 + rl) * BTS + hf * 16]);
      accA[nt] = wmma_bf16(af, ba, accA[nt]);
      bf16x16 bg = load_b_frag(&lBg[(nt * 16 + rl) * BTS + hf * 16]);
      accG[nt] = wmma_bf16(af, bg, accG[nt]);
    }
  }
#pragma unroll
  for (int nt = 0; nt < 8; nt++) {
    int n = n0 + nt * 16 + rl;
    if (n >= Hn) continue;
    float ba = bias ? bias[(size_t)z * sBiasE + n] : 0.f;
    float bg = bias ? bias[(size_t)z * sBiasE + Hn + n] : 0.f;
    float mu = mult ? mult[(size_t)z * sMultE + n] : 1.f;
#pragma unroll
    for (int vr = 0; vr < 8; vr++) {
      int m = m0 + wv * 16 + hf * 8 + vr;
      float av = accA[nt][vr] + ba;
      float gv = accG[nt][vr] + bg;
      outH[(size_t)z * sHe + (size_t)m * Hn + n] = (bf16)(av * gelu_tanh(gv) * mu);
    }
  }
}

// ------------------------------------------------------ flash attention
__global__ __launch_bounds__(256)
void k_flash(const bf16* __restrict__ q, const bf16* __restrict__ kk,
             const bf16* __restrict__ vv, const float* __restrict__ mask,
             bf16* __restrict__ ctx) {
  __shared__ __align__(16) bf16 lK[64 * KTS];
  __shared__ __align__(16) bf16 lVt[64 * KTS];
  __shared__ __align__(16) bf16 lP[8 * 16 * KTS];
  const int tid = threadIdx.x, wv = tid >> 5, ln = tid & 31;
  const int rl = ln & 15, hf = ln >> 4;
  const int qt = blockIdx.x, hh = blockIdx.y, bb = blockIdx.z;
  const int qrow = qt * 128 + wv * 16 + rl;
  const bf16* qp = q + ((size_t)(bb * SEQ + qrow)) * DMODEL + hh * HDIM;
  bf16x16 qa[2];
#pragma unroll
  for (int f = 0; f < 2; f++) qa[f] = load_a_frag(qp + f * 32 + hf * 16);
  float mrow[8], lrow[8];
  f32x8 o[4];
#pragma unroll
  for (int i = 0; i < 8; i++) { mrow[i] = -1e30f; lrow[i] = 0.f; }
#pragma unroll
  for (int i = 0; i < 4; i++) o[i] = zero8();
  bf16* pw = &lP[wv * 16 * KTS];

  for (int kt = 0; kt < SEQ / 64; kt++) {
    __syncthreads();
#pragma unroll
    for (int i = 0; i < 2; i++) {                // K,V tiles, 16B reads
      int cid = tid + i * 256;                   // 0..511
      int r = cid >> 3, c = (cid & 7) * 8;
      size_t gi = ((size_t)(bb * SEQ + kt * 64 + r)) * DMODEL + hh * HDIM + c;
      *(bf16x8*)&lK[r * KTS + c] = *(const bf16x8*)(kk + gi);
      bf16x8 v8 = *(const bf16x8*)(vv + gi);
#pragma unroll
      for (int j = 0; j < 8; j++) lVt[(c + j) * KTS + r] = v8[j];
    }
    __syncthreads();
    f32x8 s[4];
#pragma unroll
    for (int nt = 0; nt < 4; nt++) {
      s[nt] = zero8();
#pragma unroll
      for (int f = 0; f < 2; f++) {
        bf16x16 bk = load_b_frag(&lK[(nt * 16 + rl) * KTS + f * 32 + hf * 16]);
        s[nt] = wmma_bf16(qa[f], bk, s[nt]);
      }
      float mk = mask[bb * SEQ + kt * 64 + nt * 16 + rl];
#pragma unroll
      for (int vr = 0; vr < 8; vr++) s[nt][vr] += mk;
    }
    // online softmax per C-layout row
#pragma unroll
    for (int vr = 0; vr < 8; vr++) {
      float t = fmaxf(fmaxf(s[0][vr], s[1][vr]), fmaxf(s[2][vr], s[3][vr]));
      t = fmaxf(t, __shfl_xor(t, 1, 16));
      t = fmaxf(t, __shfl_xor(t, 2, 16));
      t = fmaxf(t, __shfl_xor(t, 4, 16));
      t = fmaxf(t, __shfl_xor(t, 8, 16));
      float mnew = fmaxf(mrow[vr], t);
      float alpha = __expf(mrow[vr] - mnew);
      float rs = 0.f;
#pragma unroll
      for (int nt = 0; nt < 4; nt++) {
        float pv = __expf(s[nt][vr] - mnew);
        s[nt][vr] = pv;
        rs += pv;
      }
      rs += __shfl_xor(rs, 1, 16);
      rs += __shfl_xor(rs, 2, 16);
      rs += __shfl_xor(rs, 4, 16);
      rs += __shfl_xor(rs, 8, 16);
      lrow[vr] = lrow[vr] * alpha + rs;
      mrow[vr] = mnew;
#pragma unroll
      for (int nt = 0; nt < 4; nt++) o[nt][vr] *= alpha;
    }
    // stage P (C layout -> LDS) then re-read in A-fragment layout
#pragma unroll
    for (int nt = 0; nt < 4; nt++)
#pragma unroll
      for (int vr = 0; vr < 8; vr++)
        pw[(hf * 8 + vr) * KTS + nt * 16 + rl] = (bf16)s[nt][vr];
    bf16x16 pa[2];
#pragma unroll
    for (int f = 0; f < 2; f++) pa[f] = load_a_frag(&pw[rl * KTS + f * 32 + hf * 16]);
#pragma unroll
    for (int nt = 0; nt < 4; nt++) {
#pragma unroll
      for (int f = 0; f < 2; f++) {
        bf16x16 vf = load_b_frag(&lVt[(nt * 16 + rl) * KTS + f * 32 + hf * 16]);
        o[nt] = wmma_bf16(pa[f], vf, o[nt]);
      }
    }
  }
#pragma unroll
  for (int nt = 0; nt < 4; nt++) {
#pragma unroll
    for (int vr = 0; vr < 8; vr++) {
      int m = qt * 128 + wv * 16 + hf * 8 + vr;
      float val = o[nt][vr] / lrow[vr];
      ctx[((size_t)(bb * SEQ + m)) * DMODEL + hh * HDIM + nt * 16 + rl] = (bf16)val;
    }
  }
}

// ------------------------------------------------------------- gating
__global__ __launch_bounds__(256)
void k_gating(const float* __restrict__ moein, const float* __restrict__ gw,
              int* __restrict__ tE1, int* __restrict__ tE2,
              int* __restrict__ tP1, int* __restrict__ tP2,
              float* __restrict__ tG1, float* __restrict__ tG2,
              int* __restrict__ slot, float* __restrict__ balp,
              float* __restrict__ zp) {
  __shared__ short se1[SEQ], se2[SEQ], sr2[SEQ], sp1[SEQ], sp2[SEQ];
  __shared__ float sg1[SEQ], sg2[SEQ];
  __shared__ float prd[16 * 256];
  __shared__ float zrd[256];
  const int b = blockIdx.x, tid = threadIdx.x;
  float pprox[16];
#pragma unroll
  for (int e = 0; e < 16; e++) pprox[e] = 0.f;
  float zacc = 0.f;
  for (int g = tid; g < SEQ; g += 256) {
    const float* xr = moein + ((size_t)(b * SEQ + g)) * DMODEL;
    float lg[16];
#pragma unroll
    for (int e = 0; e < 16; e++) lg[e] = 0.f;
    for (int d = 0; d < DMODEL; d++) {
      float xv = xr[d];
      const float* gr = gw + d * 16;
#pragma unroll
      for (int e = 0; e < 16; e++) lg[e] += xv * gr[e];
    }
    float mx = lg[0];
#pragma unroll
    for (int e = 1; e < 16; e++) mx = fmaxf(mx, lg[e]);
    float se = 0.f, pr[16];
#pragma unroll
    for (int e = 0; e < 16; e++) { pr[e] = __expf(lg[e] - mx); se += pr[e]; }
    float inv = 1.f / se;
#pragma unroll
    for (int e = 0; e < 16; e++) { pr[e] *= inv; pprox[e] += pr[e]; }
    float zv = mx + __logf(se);
    zacc += zv * zv;
    int i1 = 0; float p1v = pr[0];
#pragma unroll
    for (int e = 1; e < 16; e++) if (pr[e] > p1v) { p1v = pr[e]; i1 = e; }
    int i2 = -1; float p2v = -1.f;
#pragma unroll
    for (int e = 0; e < 16; e++)
      if (e != i1 && pr[e] > p2v) { p2v = pr[e]; i2 = e; }
    float den = fmaxf(p1v + p2v, 1e-9f);
    float g1 = p1v / den, g2 = p2v / den;
    se1[g] = (short)i1; se2[g] = (short)i2;
    sg1[g] = g1; sg2[g] = g2;
    sr2[g] = (g2 > 0.2f) ? 1 : 0;
  }
#pragma unroll
  for (int e = 0; e < 16; e++) prd[e * 256 + tid] = pprox[e];
  zrd[tid] = zacc;
  __syncthreads();
  for (int st = 128; st > 0; st >>= 1) {
    if (tid < st) {
#pragma unroll
      for (int e = 0; e < 16; e++) prd[e * 256 + tid] += prd[e * 256 + tid + st];
      zrd[tid] += zrd[tid + st];
    }
    __syncthreads();
  }
  if (tid == 0) {
    for (int e = 0; e < 16; e++)
      for (int c = 0; c < CAP; c++) slot[e * (BSZ * CAP) + b * CAP + c] = -1;
    int cnt[16];
    for (int e = 0; e < 16; e++) cnt[e] = 0;
    for (int g = 0; g < SEQ; g++) {          // top-1 capacity scan (in order)
      int e = se1[g];
      int p = cnt[e]++;
      if (p < CAP) { sp1[g] = (short)p; slot[e * (BSZ * CAP) + b * CAP + p] = g; }
      else { sp1[g] = -1; sg1[g] = 0.f; }
    }
    int cnt2[16];
    for (int e = 0; e < 16; e++) cnt2[e] = (cnt[e] < CAP) ? cnt[e] : CAP;
    for (int g = 0; g < SEQ; g++) {          // top-2 scan, offset by count1
      if (sr2[g]) {
        int e = se2[g];
        int p = cnt2[e]++;
        if (p < CAP) { sp2[g] = (short)p; slot[e * (BSZ * CAP) + b * CAP + p] = g; }
        else { sp2[g] = -1; sg2[g] = 0.f; }
      } else { sp2[g] = -1; sg2[g] = 0.f; }
    }
    float bal = 0.f;
    for (int e = 0; e < 16; e++) bal += (float)cnt[e] * prd[e * 256];
    balp[b] = bal / (1024.f * 1024.f);
    zp[b] = zrd[0];
  }
  __syncthreads();
  for (int g = tid; g < SEQ; g += 256) {
    size_t t = (size_t)b * SEQ + g;
    tE1[t] = se1[g]; tE2[t] = se2[g];
    tP1[t] = sp1[g]; tP2[t] = sp2[g];
    tG1[t] = sg1[g]; tG2[t] = sg2[g];
  }
}

// ------------------------------------------------------------ dispatch
__global__ __launch_bounds__(256)
void k_gather(const int* __restrict__ slot, const bf16* __restrict__ moeh,
              bf16* __restrict__ xin) {
  int row = blockIdx.x;               // e*1024 + b*128 + c
  int g = slot[row];
  int b = (row >> 7) & 7;
  bf16x4* dst = (bf16x4*)(xin + (size_t)row * DMODEL);
  int tid = threadIdx.x;
  if (g >= 0) {
    const bf16x4* src = (const bf16x4*)(moeh + ((size_t)(b * SEQ + g)) * DMODEL);
    dst[tid] = src[tid];
  } else {
    bf16x4 z4;
    z4[0] = z4[1] = z4[2] = z4[3] = (bf16)0.f;
    dst[tid] = z4;
  }
}

// ------------------------------------------------------------- combine
__global__ __launch_bounds__(256)
void k_combine(const float* __restrict__ x, const float* __restrict__ eout,
               const int* __restrict__ tE1, const int* __restrict__ tE2,
               const int* __restrict__ tP1, const int* __restrict__ tP2,
               const float* __restrict__ tG1, const float* __restrict__ tG2,
               float* __restrict__ xmoe) {
  int t = blockIdx.x;
  int b = t >> 10;
  int tid = threadIdx.x;
  int e1 = tE1[t], p1 = tP1[t], e2 = tE2[t], p2 = tP2[t];
  float g1 = tG1[t], g2 = tG2[t];
  float4 v = ((const float4*)(x + (size_t)t * DMODEL))[tid];
  if (p1 >= 0) {
    float4 a = ((const float4*)(eout +
        ((size_t)(e1 * (BSZ * CAP) + b * CAP + p1)) * DMODEL))[tid];
    v.x += g1 * a.x; v.y += g1 * a.y; v.z += g1 * a.z; v.w += g1 * a.w;
  }
  if (p2 >= 0) {
    float4 a = ((const float4*)(eout +
        ((size_t)(e2 * (BSZ * CAP) + b * CAP + p2)) * DMODEL))[tid];
    v.x += g2 * a.x; v.y += g2 * a.y; v.z += g2 * a.z; v.w += g2 * a.w;
  }
  ((float4*)(xmoe + (size_t)t * DMODEL))[tid] = v;
}

__global__ void k_aux(const float* __restrict__ balp, const float* __restrict__ zp,
                      float* __restrict__ out) {
  if (blockIdx.x == 0 && threadIdx.x == 0) {
    float bs = 0.f, zs = 0.f;
    for (int b = 0; b < BSZ; b++) { bs += balp[b]; zs += zp[b]; }
    float bal_loss = bs / (float)(BSZ * NEXP) * (float)(NEXP * NEXP);
    float z_loss = zs / (float)(BSZ * SEQ);
    out[0] = 0.01f * bal_loss + 0.001f * z_loss;
  }
}

// ================================================================== host
extern "C" void kernel_launch(void* const* d_in, const int* in_sizes, int n_in,
                              void* d_out, int out_size, void* d_ws, size_t ws_size,
                              hipStream_t stream) {
  (void)in_sizes; (void)n_in; (void)out_size; (void)ws_size;
  const float* hs    = (const float*)d_in[0];
  const float* mask  = (const float*)d_in[1];
  const float* lnaw  = (const float*)d_in[2];
  const float* wq    = (const float*)d_in[3];
  const float* wk    = (const float*)d_in[4];
  const float* wvv   = (const float*)d_in[5];
  const float* wo    = (const float*)d_in[6];
  const float* lnfw  = (const float*)d_in[7];
  const float* moenw = (const float*)d_in[8];
  const float* gatew = (const float*)d_in[9];
  const float* ewin  = (const float*)d_in[10];
  const float* ebin  = (const float*)d_in[11];
  const float* emult = (const float*)d_in[12];
  const float* ewout = (const float*)d_in[13];
  const float* ebout = (const float*)d_in[14];
  const float* ffnw  = (const float*)d_in[15];
  const float* ffw1  = (const float*)d_in[16];
  const float* ffb1  = (const float*)d_in[17];
  const float* ffw2  = (const float*)d_in[18];
  const float* ffb2  = (const float*)d_in[19];
  float* out = (float*)d_out;

  char* base = (char*)d_ws;
  size_t off = 0;
  auto alloc = [&](size_t bytes) -> void* {
    void* p = base + off;
    off = (off + bytes + 255) & ~(size_t)255;
    return p;
  };
  bf16* wq_h   = (bf16*)alloc((size_t)DMODEL * DMODEL * 2);
  bf16* wk_h   = (bf16*)alloc((size_t)DMODEL * DMODEL * 2);
  bf16* wv_h   = (bf16*)alloc((size_t)DMODEL * DMODEL * 2);
  bf16* wo_h   = (bf16*)alloc((size_t)DMODEL * DMODEL * 2);
  bf16* ewin_h = (bf16*)alloc((size_t)NEXP * DMODEL * 2 * DHID * 2);
  bf16* ewout_h= (bf16*)alloc((size_t)NEXP * DHID * DMODEL * 2);
  bf16* ffw1_h = (bf16*)alloc((size_t)DMODEL * 2 * DHID * 2);
  bf16* ffw2_h = (bf16*)alloc((size_t)DHID * DMODEL * 2);
  bf16* norm_h = (bf16*)alloc((size_t)NTOK * DMODEL * 2);
  bf16* q_h    = (bf16*)alloc((size_t)NTOK * DMODEL * 2);
  bf16* k_h    = (bf16*)alloc((size_t)NTOK * DMODEL * 2);
  bf16* v_h    = (bf16*)alloc((size_t)NTOK * DMODEL * 2);
  bf16* ctx_h  = (bf16*)alloc((size_t)NTOK * DMODEL * 2);
  float* hidden= (float*)alloc((size_t)NTOK * DMODEL * 4);
  float* xln   = (float*)alloc((size_t)NTOK * DMODEL * 4);
  float* moein_f = (float*)alloc((size_t)NTOK * DMODEL * 4);
  bf16* moein_h  = (bf16*)alloc((size_t)NTOK * DMODEL * 2);
  int* tE1 = (int*)alloc((size_t)NTOK * 4);
  int* tE2 = (int*)alloc((size_t)NTOK * 4);
  int* tP1 = (int*)alloc((size_t)NTOK * 4);
  int* tP2 = (int*)alloc((size_t)NTOK * 4);
  float* tG1 = (float*)alloc((size_t)NTOK * 4);
  float* tG2 = (float*)alloc((size_t)NTOK * 4);
  int* slot  = (int*)alloc((size_t)NEXP * BSZ * CAP * 4);
  float* balp  = (float*)alloc(BSZ * 4);
  float* zpart = (float*)alloc(BSZ * 4);
  bf16* xin_h = (bf16*)alloc((size_t)NEXP * BSZ * CAP * DMODEL * 2);
  bf16* hhe   = (bf16*)alloc((size_t)NEXP * BSZ * CAP * DHID * 2);
  float* eout = (float*)alloc((size_t)NEXP * BSZ * CAP * DMODEL * 4);
  float* xmoe = (float*)alloc((size_t)NTOK * DMODEL * 4);
  bf16* fx_h  = (bf16*)alloc((size_t)NTOK * DMODEL * 2);
  bf16* hhf   = (bf16*)alloc((size_t)NTOK * DHID * 2);

  auto cv = [&](const float* s, bf16* dp, long long n) {
    int blocks = (int)((n / 4 + 255) / 256);
    if (blocks > 16384) blocks = 16384;
    if (blocks < 1) blocks = 1;
    k_f32_to_bf16<<<blocks, 256, 0, stream>>>(s, dp, n);
  };
  cv(wq, wq_h, (long long)DMODEL * DMODEL);
  cv(wk, wk_h, (long long)DMODEL * DMODEL);
  cv(wvv, wv_h, (long long)DMODEL * DMODEL);
  cv(wo, wo_h, (long long)DMODEL * DMODEL);
  cv(ewin, ewin_h, (long long)NEXP * DMODEL * 2 * DHID);
  cv(ewout, ewout_h, (long long)NEXP * DHID * DMODEL);
  cv(ffw1, ffw1_h, (long long)DMODEL * 2 * DHID);
  cv(ffw2, ffw2_h, (long long)DHID * DMODEL);

  // attention sub-layer
  k_norm<<<NTOK, 256, 0, stream>>>(hs, lnaw, nullptr, norm_h, 0);
  dim3 gQ(DMODEL / 128, NTOK / 128, 1);
  k_gemm<<<gQ, 256, 0, stream>>>(norm_h, 0, DMODEL, wq_h, 0, DMODEL, nullptr, 0,
                                 nullptr, nullptr, nullptr, q_h, 0,
                                 NTOK, DMODEL, DMODEL, DMODEL);
  k_gemm<<<gQ, 256, 0, stream>>>(norm_h, 0, DMODEL, wk_h, 0, DMODEL, nullptr, 0,
                                 nullptr, nullptr, nullptr, k_h, 0,
                                 NTOK, DMODEL, DMODEL, DMODEL);
  k_gemm<<<gQ, 256, 0, stream>>>(norm_h, 0, DMODEL, wv_h, 0, DMODEL, nullptr, 0,
                                 nullptr, nullptr, nullptr, v_h, 0,
                                 NTOK, DMODEL, DMODEL, DMODEL);
  k_flash<<<dim3(SEQ / 128, NHEADS, BSZ), 256, 0, stream>>>(q_h, k_h, v_h, mask, ctx_h);
  k_gemm<<<gQ, 256, 0, stream>>>(ctx_h, 0, DMODEL, wo_h, 0, DMODEL, nullptr, 0,
                                 hs, nullptr, hidden, nullptr, 0,
                                 NTOK, DMODEL, DMODEL, DMODEL);
  // norms + gating + dispatch
  k_norm<<<NTOK, 256, 0, stream>>>(hidden, lnfw, xln, nullptr, 0);
  k_norm<<<NTOK, 256, 0, stream>>>(xln, moenw, moein_f, moein_h, 1);
  k_gating<<<BSZ, 256, 0, stream>>>(moein_f, gatew, tE1, tE2, tP1, tP2, tG1, tG2,
                                    slot, balp, zpart);
  k_gather<<<NEXP * BSZ * CAP, 256, 0, stream>>>(slot, moein_h, xin_h);
  // experts
  dim3 gE1((DHID + 127) / 128, (BSZ * CAP) / 128, NEXP);
  k_gemm_glu<<<gE1, 256, 0, stream>>>(
      xin_h, (long long)BSZ * CAP * DMODEL, DMODEL,
      ewin_h, (long long)DMODEL * 2 * DHID, 2 * DHID,
      ebin, 2 * DHID, emult, DHID,
      hhe, (long long)BSZ * CAP * DHID, BSZ * CAP, DHID, DMODEL);
  dim3 gE2(DMODEL / 128, (BSZ * CAP) / 128, NEXP);
  k_gemm<<<gE2, 256, 0, stream>>>(
      hhe, (long long)BSZ * CAP * DHID, DHID,
      ewout_h, (long long)DHID * DMODEL, DMODEL,
      ebout, DMODEL, nullptr, nullptr,
      eout, nullptr, (long long)BSZ * CAP * DMODEL,
      BSZ * CAP, DMODEL, DHID, DMODEL);
  k_combine<<<NTOK, 256, 0, stream>>>(xln, eout, tE1, tE2, tP1, tP2, tG1, tG2, xmoe);
  // ff_after + final residuals into d_out
  k_norm<<<NTOK, 256, 0, stream>>>(xmoe, ffnw, nullptr, fx_h, 1);
  dim3 gF1((DHID + 127) / 128, NTOK / 128, 1);
  k_gemm_glu<<<gF1, 256, 0, stream>>>(fx_h, 0, DMODEL, ffw1_h, 0, 2 * DHID,
                                      ffb1, 0, nullptr, 0, hhf, 0,
                                      NTOK, DHID, DMODEL);
  dim3 gF2(DMODEL / 128, NTOK / 128, 1);
  k_gemm<<<gF2, 256, 0, stream>>>(hhf, 0, DHID, ffw2_h, 0, DMODEL, ffb2, 0,
                                  hidden, xmoe, out, nullptr, 0,
                                  NTOK, DMODEL, DHID, DMODEL);
  k_aux<<<1, 64, 0, stream>>>(balp, zpart, out + (size_t)NTOK * DMODEL);
}
